// IQCA_8134668058840
// MI455X (gfx1250) — matrix operand order repaired and checked by reading, not compiled, not run on hardware
//
#include <hip/hip_runtime.h>
#include <hip/hip_bf16.h>
#include <math.h>

// Problem constants (match reference)
#define CCH   64          // channels
#define HWSZ  4096        // H*W
#define NBAT  4           // batch
#define FELT  (NBAT*CCH*HWSZ)   // 1,048,576 elements per (B,C,H,W) tensor

typedef __attribute__((ext_vector_type(16))) __bf16 bf16x16;
typedef __attribute__((ext_vector_type(8)))  __bf16 bf16x8;
typedef __attribute__((ext_vector_type(8)))  float  f32x8;
typedef __attribute__((ext_vector_type(4)))  float  f32x4;
typedef __attribute__((ext_vector_type(4)))  int    i32x4v;

#if __has_builtin(__builtin_amdgcn_global_load_async_to_lds_b128)
#define HAVE_ASYNC_LDS 1
#else
#define HAVE_ASYNC_LDS 0
#endif

static __device__ __forceinline__ f32x8 wmma_bf16(bf16x16 a, bf16x16 b, f32x8 c) {
  // D(16x16,f32) = A(16x32,bf16) x B(32x16,bf16) + C
  return __builtin_amdgcn_wmma_f32_16x16x32_bf16(false, a, false, b, (short)0, c,
                                                 false, false);
}
static __device__ __forceinline__ f32x8 zero8() {
  f32x8 v;
#pragma unroll
  for (int i = 0; i < 8; ++i) v[i] = 0.0f;
  return v;
}
// Convert 8 consecutive f32 (32B aligned) to bf16x8.
static __device__ __forceinline__ bf16x8 cvt8(const float* __restrict__ src) {
  f32x4 a0 = ((const f32x4*)src)[0];
  f32x4 a1 = ((const f32x4*)src)[1];
  bf16x8 v;
#pragma unroll
  for (int j = 0; j < 4; ++j) { v[j] = (__bf16)a0[j]; v[4 + j] = (__bf16)a1[j]; }
  return v;
}

// ---------------------------------------------------------------------------
// LayerNorm over channel axis: per (b, n) normalize 64 channels.
// ---------------------------------------------------------------------------
__global__ __launch_bounds__(256) void cdna5_ln_channel(
    const float* __restrict__ x, const float* __restrict__ gam,
    const float* __restrict__ bet, float* __restrict__ out) {
  int i = blockIdx.x * 256 + threadIdx.x;          // over B*HW
  int bi = i >> 12, n = i & (HWSZ - 1);
  const float* xp = x + ((size_t)bi * CCH) * HWSZ + n;
  float s = 0.f, ss = 0.f;
#pragma unroll 8
  for (int c = 0; c < CCH; ++c) { float v = xp[(size_t)c * HWSZ]; s += v; ss += v * v; }
  float mu = s * (1.0f / CCH);
  float var = ss * (1.0f / CCH) - mu * mu;
  float rstd = rsqrtf(var + 1e-5f);
  float* op = out + ((size_t)bi * CCH) * HWSZ + n;
#pragma unroll 8
  for (int c = 0; c < CCH; ++c) {
    float v = xp[(size_t)c * HWSZ];
    op[(size_t)c * HWSZ] = (v - mu) * rstd * gam[c] + bet[c];
  }
}

// ---------------------------------------------------------------------------
// Pointwise (1x1) conv as WMMA GEMM: out[b,co,n] = sum_ci W[co,ci]*in[b,ci,n]+bias
// Block tile: 64(cout) x 64(n); 8 waves -> each wave 16x32. K fully unrolled.
// LDS: W tile [64][CIN] (row-major, A-frag contiguous) +
//      input tile TRANSPOSED [n(64)][CIN] so B-frags are K-contiguous b128s.
// ---------------------------------------------------------------------------
template <int CIN, int COUT>
__global__ __launch_bounds__(256) void cdna5_pw_conv_wmma(
    const float* __restrict__ in, const float* __restrict__ w,
    const float* __restrict__ bias, const float* __restrict__ residual,
    float* __restrict__ out) {
  extern __shared__ char smemraw[];
  __bf16* lw  = (__bf16*)smemraw;       // [64][CIN]
  __bf16* lit = lw + 64 * CIN;          // [64 n][CIN]   (transposed input tile)
  const int tid = threadIdx.x;
  const int b = blockIdx.z;
  const int n0 = blockIdx.x * 64;
  const int m0 = blockIdx.y * 64;
  // W tile: rows contiguous in global; 8 elems per thread, b128 load + b128 store
#pragma unroll
  for (int base = tid * 8; base < 64 * CIN; base += 256 * 8) {
    int co = base / CIN, ci = base % CIN;
    *(bf16x8*)(lw + base) = cvt8(&w[(size_t)(m0 + co) * CIN + ci]);
  }
  // input tile, transposed: read 8 contiguous n (coalesced), scatter into [n][ci]
#pragma unroll
  for (int g = tid; g < CIN * 8; g += 256) {
    int ci = g >> 3, n = (g & 7) * 8;
    bf16x8 v = cvt8(&in[((size_t)b * CIN + ci) * HWSZ + n0 + n]);
#pragma unroll
    for (int j = 0; j < 8; ++j) lit[(n + j) * CIN + ci] = v[j];
  }
  __syncthreads();
  const int lane = tid & 31, wave = tid >> 5;
  const int mw = (wave & 3) * 16, nw = (wave >> 2) * 32;
  const int mrow = lane & 15, kb = (lane >> 4) * 8;
  const int khalf = (lane >> 4) * 16, ncol = lane & 15;
  const __bf16* lirow0 = lit + (nw + ncol) * CIN;
  const __bf16* lirow1 = lit + (nw + 16 + ncol) * CIN;
  f32x8 acc0 = zero8(), acc1 = zero8();
#pragma unroll
  for (int ks = 0; ks < CIN; ks += 32) {
    bf16x16 a;
    const bf16x8* ap = (const bf16x8*)(lw + (mw + mrow) * CIN + ks + khalf);
    bf16x8 alo = ap[0], ahi = ap[1];
#pragma unroll
    for (int e = 0; e < 8; ++e) { a[e] = alo[e]; a[8 + e] = ahi[e]; }
    bf16x16 b0 = *(const bf16x16*)(lirow0 + ks + khalf);
    bf16x16 b1 = *(const bf16x16*)(lirow1 + ks + khalf);
    acc0 = wmma_bf16(a, b0, acc0);
    acc1 = wmma_bf16(a, b1, acc1);
  }
#pragma unroll
  for (int r = 0; r < 8; ++r) {
    int co = m0 + mw + (lane >> 4) * 8 + r;
    int n = n0 + nw + ncol;
    size_t o0 = ((size_t)b * COUT + co) * HWSZ + n;
    float bi = bias[co];
    float v0 = acc0[r] + bi, v1 = acc1[r] + bi;
    if (residual) { v0 += residual[o0]; v1 += residual[o0 + 16]; }
    out[o0] = v0; out[o0 + 16] = v1;
  }
}

// ---------------------------------------------------------------------------
// Depthwise 3x3 SAME conv (64x64 spatial), one thread per output element.
// ---------------------------------------------------------------------------
__global__ __launch_bounds__(256) void cdna5_dw3x3(
    const float* __restrict__ in, const float* __restrict__ wd,
    const float* __restrict__ bd, float* __restrict__ out) {
  int i = blockIdx.x * 256 + threadIdx.x;          // B*C*HW
  int n = i & (HWSZ - 1); int c = (i >> 12) & (CCH - 1); int b = i >> 18;
  int yy = n >> 6, xx = n & 63;
  const float* ip = in + ((size_t)b * CCH + c) * HWSZ;
  const float* wp = wd + c * 9;
  float acc = bd[c];
#pragma unroll
  for (int ky = 0; ky < 3; ++ky) {
    int y = yy + ky - 1;
    if (y < 0 || y > 63) continue;
#pragma unroll
    for (int kx = 0; kx < 3; ++kx) {
      int x = xx + kx - 1;
      if (x < 0 || x > 63) continue;
      acc += ip[y * 64 + x] * wp[ky * 3 + kx];
    }
  }
  out[i] = acc;
}

// Fill a [c][n] bf16 LDS tile (row-major, n contiguous) from f32 [b][c][n].
template <int WID>
static __device__ __forceinline__ void fill_tile(
    __bf16* dst, const float* __restrict__ src_base, int b, int nbase, int tid) {
#pragma unroll
  for (int base = tid * 8; base < CCH * WID; base += 256 * 8) {
    int c = base / WID, n = base % WID;
    const float* src = &src_base[((size_t)b * CCH + c) * HWSZ + nbase + n];
    __builtin_prefetch(src + WID, 0, 0);
    *(bf16x8*)(dst + base) = cvt8(src);
  }
}
// Fill a TRANSPOSED [m(64)][c(CCH)] bf16 LDS tile from f32 [b][c][m].
static __device__ __forceinline__ void fill_tile_t(
    __bf16* dst, const float* __restrict__ src_base, int b, int mbase, int tid) {
#pragma unroll
  for (int g = tid; g < CCH * 8; g += 256) {
    int c = g >> 3, m = (g & 7) * 8;
    const float* src = &src_base[((size_t)b * CCH + c) * HWSZ + mbase + m];
    __builtin_prefetch(src + 64, 0, 0);
    bf16x8 v = cvt8(src);
#pragma unroll
    for (int j = 0; j < 8; ++j) dst[(m + j) * CCH + c] = v[j];
  }
}

// ---------------------------------------------------------------------------
// Attention pass 1: S = q^T k via WMMA, online row max / row sum of exp.
// Block: 128 rows of n (8 waves x 16 rows), streams all 4096 m in tiles of 64.
// K tile stored transposed [m][c] so S B-frags are K-contiguous b128 loads.
// ---------------------------------------------------------------------------
__global__ __launch_bounds__(256) void cdna5_attn_stats(
    const float* __restrict__ qf, const float* __restrict__ kf,
    float* __restrict__ rowmax, float* __restrict__ rowsum) {
  __shared__ __bf16 lq[CCH * 128];    // [c][n]  16KB
  __shared__ __bf16 lkt[64 * CCH];    // [m][c]   8KB (transposed)
  const int tid = threadIdx.x, lane = tid & 31, wave = tid >> 5;
  const int b = blockIdx.y, n0 = blockIdx.x * 128;
  fill_tile<128>(lq, qf, b, n0, tid);
  __syncthreads();
  const int mrow = lane & 15, kb = (lane >> 4) * 8;
  const int khalf = (lane >> 4) * 16, ncol = lane & 15;
  const int nw = wave * 16;
  bf16x16 aq[2];
#pragma unroll
  for (int ks = 0; ks < 2; ++ks)
#pragma unroll
    for (int e = 0; e < 16; ++e) {
      int k = ks * 32 + khalf + e;
      aq[ks][e] = lq[k * 128 + nw + mrow];
    }
  float mrun[8], srun[8];
#pragma unroll
  for (int r = 0; r < 8; ++r) { mrun[r] = -3.0e38f; srun[r] = 0.f; }
  for (int mt = 0; mt < HWSZ; mt += 64) {
    __syncthreads();
    fill_tile_t(lkt, kf, b, mt, tid);
    __syncthreads();
    f32x8 s[4];
#pragma unroll
    for (int ms = 0; ms < 4; ++ms) {
      s[ms] = zero8();
#pragma unroll
      for (int ks = 0; ks < 2; ++ks) {
        bf16x16 bk = *(const bf16x16*)(lkt + (ms * 16 + ncol) * CCH + ks * 32 + khalf);
        s[ms] = wmma_bf16(aq[ks], bk, s[ms]);
      }
    }
#pragma unroll
    for (int r = 0; r < 8; ++r) {
      float mx = fmaxf(fmaxf(s[0][r], s[1][r]), fmaxf(s[2][r], s[3][r]));
#pragma unroll
      for (int off = 1; off < 16; off <<= 1) mx = fmaxf(mx, __shfl_xor(mx, off, 32));
      float nm = fmaxf(mrun[r], mx);
      float ps = __expf(s[0][r] - nm) + __expf(s[1][r] - nm) +
                 __expf(s[2][r] - nm) + __expf(s[3][r] - nm);
#pragma unroll
      for (int off = 1; off < 16; off <<= 1) ps += __shfl_xor(ps, off, 32);
      srun[r] = srun[r] * __expf(mrun[r] - nm) + ps;
      mrun[r] = nm;
    }
  }
  if ((lane & 15) == 0) {
#pragma unroll
    for (int r = 0; r < 8; ++r) {
      int n = n0 + nw + (lane >> 4) * 8 + r;
      rowmax[b * HWSZ + n] = mrun[r];
      rowsum[b * HWSZ + n] = srun[r];
    }
  }
}

// ---------------------------------------------------------------------------
// Attention pass 2: recompute S, P = exp(S-max)/(64*sum), O^T = P V^T (WMMA).
// Writes oT2 in [b][c][m] layout (bf16) so the proj GEMM's LDS tile is a
// pure contiguous copy AND K-contiguous for its B-frags.
// ---------------------------------------------------------------------------
__global__ __launch_bounds__(256) void cdna5_attn_av(
    const float* __restrict__ qf, const float* __restrict__ kf,
    const float* __restrict__ vf, const float* __restrict__ rowmax,
    const float* __restrict__ rowsum, __bf16* __restrict__ oT2) {
  __shared__ __bf16 lq[CCH * 128];      // [c][n] 16KB
  __shared__ __bf16 lkt[64 * CCH];      // [m][c]  8KB (transposed)
  __shared__ __bf16 lv[CCH * 64];       // [c][m]  8KB (K-contiguous for PV)
  __shared__ __bf16 lp[8][16 * 64];     // per-wave P relayout slab, 16KB
  const int tid = threadIdx.x, lane = tid & 31, wave = tid >> 5;
  const int b = blockIdx.y, n0 = blockIdx.x * 128;
  fill_tile<128>(lq, qf, b, n0, tid);
  __syncthreads();
  const int mrow = lane & 15, kb = (lane >> 4) * 8;
  const int khalf = (lane >> 4) * 16, ncol = lane & 15;
  const int nw = wave * 16;
  bf16x16 aq[2];
#pragma unroll
  for (int ks = 0; ks < 2; ++ks)
#pragma unroll
    for (int e = 0; e < 16; ++e) {
      int k = ks * 32 + khalf + e;
      aq[ks][e] = lq[k * 128 + nw + mrow];
    }
  float rmax[8], rscl[8];
#pragma unroll
  for (int r = 0; r < 8; ++r) {
    int n = n0 + nw + (lane >> 4) * 8 + r;
    rmax[r] = rowmax[b * HWSZ + n];
    rscl[r] = 1.0f / (64.0f * rowsum[b * HWSZ + n]);   // softmax / sqrt(4096)
  }
  f32x8 o[4];
#pragma unroll
  for (int cs = 0; cs < 4; ++cs) o[cs] = zero8();
  for (int mt = 0; mt < HWSZ; mt += 64) {
    __syncthreads();
    fill_tile_t(lkt, kf, b, mt, tid);
    fill_tile<64>(lv, vf, b, mt, tid);
    __syncthreads();
#pragma unroll
    for (int ms = 0; ms < 4; ++ms) {
      f32x8 s = zero8();
#pragma unroll
      for (int ks = 0; ks < 2; ++ks) {
        bf16x16 bk = *(const bf16x16*)(lkt + (ms * 16 + ncol) * CCH + ks * 32 + khalf);
        s = wmma_bf16(aq[ks], bk, s);
      }
      // P tile in C/D layout -> wave-private LDS (D row = half*8+r, col = lane&15)
#pragma unroll
      for (int r = 0; r < 8; ++r) {
        float p = __expf(s[r] - rmax[r]);
        lp[wave][((lane >> 4) * 8 + r) * 64 + ms * 16 + ncol] = (__bf16)p;
      }
    }
    // O^T += P x V^T   (K-dim = m, 2 steps of 32)
#pragma unroll
    for (int ks = 0; ks < 2; ++ks) {
      bf16x16 ap;
      const bf16x8* pp = (const bf16x8*)(&lp[wave][mrow * 64 + ks * 32 + khalf]);
      bf16x8 plo = pp[0], phi = pp[1];
#pragma unroll
      for (int e = 0; e < 8; ++e) { ap[e] = plo[e]; ap[8 + e] = phi[e]; }
#pragma unroll
      for (int cs = 0; cs < 4; ++cs) {
        bf16x16 bv = *(const bf16x16*)(lv + (cs * 16 + ncol) * 64 + ks * 32 + khalf);
        o[cs] = wmma_bf16(ap, bv, o[cs]);
      }
    }
  }
  // store O^T into [b][c][n] layout (scattered b16, oT2 is only 2MB total)
#pragma unroll
  for (int cs = 0; cs < 4; ++cs)
#pragma unroll
    for (int r = 0; r < 8; ++r) {
      int n = n0 + nw + (lane >> 4) * 8 + r;
      int c = cs * 16 + ncol;
      oT2[((size_t)b * CCH + c) * HWSZ + n] = (__bf16)(o[cs][r] * rscl[r]);
    }
}

// ---------------------------------------------------------------------------
// Spatial proj: oproj[b][n][c] = sum_m proj_w[n][m] * oT2[b][c][m]
// M=4096(n, 128/block), K=4096, N=64(c). A from global f32 (cvt), B via LDS.
// LDS tile [c][m]: contiguous 16B copy per thread (async path) AND the
// B-frags read K-contiguous -> 2 x ds_load_b128 each.
// ---------------------------------------------------------------------------
__global__ __launch_bounds__(256) void cdna5_proj_gemm(
    const __bf16* __restrict__ oT2, const float* __restrict__ pw,
    float* __restrict__ oproj) {
  __shared__ __bf16 lo[CCH * 32];    // [c][m] tile, 4KB
  const int tid = threadIdx.x, lane = tid & 31, wave = tid >> 5;
  const int b = blockIdx.y, n0 = blockIdx.x * 128;
  const int nwave = n0 + wave * 16;
  const int mrow = lane & 15, kb = (lane >> 4) * 8;
  const int khalf = (lane >> 4) * 16, ncol = lane & 15;
  const int fc = tid >> 2, fm = (tid & 3) * 8;   // fill coords: c, m-offset
  f32x8 acc[4];
#pragma unroll
  for (int cs = 0; cs < 4; ++cs) acc[cs] = zero8();
  const float* wrow = pw + (size_t)(nwave + mrow) * HWSZ;
  for (int mt = 0; mt < HWSZ; mt += 32) {
    __syncthreads();
    {
      // [c][m] tile: each thread one 16B chunk within a row of oT2[b][c][:]
      const __bf16* src = oT2 + ((size_t)b * CCH + fc) * HWSZ + mt + fm;
      __bf16* dst = lo + fc * 32 + fm;
#if HAVE_ASYNC_LDS
      __builtin_amdgcn_global_load_async_to_lds_b128(
          (__attribute__((address_space(1))) i32x4v*)src,
          (__attribute__((address_space(3))) i32x4v*)dst, 0, 0);
#if __has_builtin(__builtin_amdgcn_s_wait_asynccnt)
      __builtin_amdgcn_s_wait_asynccnt(0);
#else
      asm volatile("s_wait_asynccnt 0" ::: "memory");
#endif
#else
      *(bf16x8*)dst = *(const bf16x8*)src;
#endif
    }
    __syncthreads();
    const float* wp = wrow + mt + khalf;
    __builtin_prefetch(wp + 64, 0, 0);
    bf16x16 a;
    f32x4 w0 = ((const f32x4*)wp)[0], w1 = ((const f32x4*)wp)[1];
    f32x4 w2 = ((const f32x4*)wp)[2], w3 = ((const f32x4*)wp)[3];
#pragma unroll
    for (int e = 0; e < 4; ++e) {
      a[e] = (__bf16)w0[e];  a[4 + e]  = (__bf16)w1[e];
      a[8 + e] = (__bf16)w2[e]; a[12 + e] = (__bf16)w3[e];
    }
#pragma unroll
    for (int cs = 0; cs < 4; ++cs) {
      bf16x16 bo = *(const bf16x16*)(lo + (cs * 16 + ncol) * 32 + khalf);
      acc[cs] = wmma_bf16(a, bo, acc[cs]);
    }
  }
#pragma unroll
  for (int cs = 0; cs < 4; ++cs)
#pragma unroll
    for (int r = 0; r < 8; ++r) {
      int n = nwave + (lane >> 4) * 8 + r;
      oproj[((size_t)b * HWSZ + n) * CCH + cs * 16 + ncol] = acc[cs][r];
    }
}

// x1[b,c,n] = x[b,c,n] + oproj[b,n,c] + proj_b[n]
__global__ __launch_bounds__(256) void cdna5_attn_res(
    const float* __restrict__ x, const float* __restrict__ oproj,
    const float* __restrict__ pb, float* __restrict__ x1) {
  int i = blockIdx.x * 256 + threadIdx.x;
  int n = i & (HWSZ - 1); int c = (i >> 12) & (CCH - 1); int b = i >> 18;
  x1[i] = x[i] + oproj[((size_t)b * HWSZ + n) * CCH + c] + pb[n];
}

// score *= gelu(gate)  (exact gelu, erf form)
__global__ __launch_bounds__(256) void cdna5_gate_mul(
    float* __restrict__ score, const float* __restrict__ gate) {
  int i = blockIdx.x * 256 + threadIdx.x;
  float g = gate[i];
  float ge = 0.5f * g * (1.0f + erff(g * 0.70710678118654752f));
  score[i] *= ge;
}

// ---------------------------------------------------------------------------
// Host-side orchestration
// ---------------------------------------------------------------------------
template <int COUT2>
static void run_triple(const float* in, const float* w1, const float* b1,
                       const float* wd, const float* bd, const float* w2,
                       const float* b2, float* tmp1, float* tmp2, float* dst,
                       hipStream_t stream) {
  dim3 blk(256);
  cdna5_pw_conv_wmma<64, 64><<<dim3(HWSZ / 64, 1, NBAT), blk, 64 * 256, stream>>>(
      in, w1, b1, nullptr, tmp1);
  cdna5_dw3x3<<<FELT / 256, blk, 0, stream>>>(tmp1, wd, bd, tmp2);
  cdna5_pw_conv_wmma<64, COUT2><<<dim3(HWSZ / 64, COUT2 / 64, NBAT), blk, 64 * 256,
                                  stream>>>(tmp2, w2, b2, nullptr, dst);
}

extern "C" void kernel_launch(void* const* d_in, const int* in_sizes, int n_in,
                              void* d_out, int out_size, void* d_ws, size_t ws_size,
                              hipStream_t stream) {
  (void)in_sizes; (void)n_in; (void)out_size; (void)ws_size;
  const float* x      = (const float*)d_in[0];
  const float* ln1_g  = (const float*)d_in[1];
  const float* ln1_b  = (const float*)d_in[2];
  const float* ln2_g  = (const float*)d_in[3];
  const float* ln2_b  = (const float*)d_in[4];
  const float* qw1 = (const float*)d_in[5],  *qb1 = (const float*)d_in[6];
  const float* qwd = (const float*)d_in[7],  *qbd = (const float*)d_in[8];
  const float* qw2 = (const float*)d_in[9],  *qb2 = (const float*)d_in[10];
  const float* kw1 = (const float*)d_in[11], *kb1 = (const float*)d_in[12];
  const float* kwd = (const float*)d_in[13], *kbd = (const float*)d_in[14];
  const float* kw2 = (const float*)d_in[15], *kb2 = (const float*)d_in[16];
  const float* vw1 = (const float*)d_in[17], *vb1 = (const float*)d_in[18];
  const float* vwd = (const float*)d_in[19], *vbd = (const float*)d_in[20];
  const float* vw2 = (const float*)d_in[21], *vb2 = (const float*)d_in[22];
  const float* sw1 = (const float*)d_in[23], *sb1 = (const float*)d_in[24];
  const float* swd = (const float*)d_in[25], *sbd = (const float*)d_in[26];
  const float* sw2 = (const float*)d_in[27], *sb2 = (const float*)d_in[28];
  const float* gw1 = (const float*)d_in[29], *gb1 = (const float*)d_in[30];
  const float* gwd = (const float*)d_in[31], *gbd = (const float*)d_in[32];
  const float* gw2 = (const float*)d_in[33], *gb2 = (const float*)d_in[34];
  const float* down_w = (const float*)d_in[35];
  const float* down_b = (const float*)d_in[36];
  const float* proj_w = (const float*)d_in[37];
  const float* proj_b = (const float*)d_in[38];
  float* out = (float*)d_out;

  // Workspace arena (~66 MB)
  float* xn    = (float*)d_ws;                 // 4MB (reused for xn2)
  float* t1    = xn + FELT;                    // 4MB
  float* t2    = t1 + FELT;                    // 4MB
  float* qf    = t2 + FELT;                    // 4MB
  float* kf    = qf + FELT;                    // 4MB
  float* vf    = kf + FELT;                    // 4MB
  float* rmx   = vf + FELT;                    // 64KB
  float* rsm   = rmx + NBAT * HWSZ;            // 64KB
  __bf16* oT2  = (__bf16*)(rsm + NBAT * HWSZ); // 2MB, [b][c][m] bf16
  float* oproj = (float*)(oT2 + FELT);         // 4MB
  float* x1    = oproj + FELT;                 // 4MB
  float* scoreb = x1 + FELT;                   // 16MB
  float* gateb  = scoreb + 4 * (size_t)FELT;   // 16MB

  dim3 blk(256);
  // ---- attention half ----
  cdna5_ln_channel<<<NBAT * HWSZ / 256, blk, 0, stream>>>(x, ln1_g, ln1_b, xn);
  run_triple<64>(xn, qw1, qb1, qwd, qbd, qw2, qb2, t1, t2, qf, stream);
  run_triple<64>(xn, kw1, kb1, kwd, kbd, kw2, kb2, t1, t2, kf, stream);
  run_triple<64>(xn, vw1, vb1, vwd, vbd, vw2, vb2, t1, t2, vf, stream);
  cdna5_attn_stats<<<dim3(HWSZ / 128, NBAT), blk, 0, stream>>>(qf, kf, rmx, rsm);
  cdna5_attn_av<<<dim3(HWSZ / 128, NBAT), blk, 0, stream>>>(qf, kf, vf, rmx, rsm, oT2);
  cdna5_proj_gemm<<<dim3(HWSZ / 128, NBAT), blk, 0, stream>>>(oT2, proj_w, oproj);
  cdna5_attn_res<<<FELT / 256, blk, 0, stream>>>(x, oproj, proj_b, x1);
  // ---- gated conv-FFN half ----
  cdna5_ln_channel<<<NBAT * HWSZ / 256, blk, 0, stream>>>(x1, ln2_g, ln2_b, xn);
  run_triple<256>(xn, sw1, sb1, swd, sbd, sw2, sb2, t1, t2, scoreb, stream);
  run_triple<256>(xn, gw1, gb1, gwd, gbd, gw2, gb2, t1, t2, gateb, stream);
  cdna5_gate_mul<<<4 * FELT / 256, blk, 0, stream>>>(scoreb, gateb);
  // down conv (256->64) with residual x1, writes final output. LDS = 64KB.
  cdna5_pw_conv_wmma<256, 64><<<dim3(HWSZ / 64, 1, NBAT), blk, 256 * 256, stream>>>(
      scoreb, down_w, down_b, x1, out);
}
